// DeepTensorNeuralNetwork_74680891342858
// MI455X (gfx1250) — compile-verified, matrix operand order checked by reading, not run on hardware
//
#include <hip/hip_runtime.h>
#include <hip/hip_bf16.h>

// Shapes (fixed by the reference)
// B=64, N=128, F=128, R=64, H=256, L=3, FC=256, OUT=1
// rows = B*N = 8192

typedef __attribute__((ext_vector_type(16))) _Float16 v16h;
typedef __attribute__((ext_vector_type(8)))  _Float16 v8h;
typedef __attribute__((ext_vector_type(8)))  float    v8f;
typedef __attribute__((ext_vector_type(4)))  float    v4f;
typedef __attribute__((ext_vector_type(4)))  unsigned int v4u;
typedef __attribute__((ext_vector_type(8)))  int      v8i;
typedef __attribute__((ext_vector_type(4)))  int      v4i;

union V16U { v16h v; v8h h[2]; };

// Load a 16x32 f16 WMMA fragment (A or B) from a row-major source:
// A: row M = row0+r of the left matrix; B: column N = row0+r of B is a
// contiguous row of the row-major right-hand source (B = src^T).
//   lanes 0-15 : row = row0+lane,    K = {k0..k0+7, k0+16..k0+23}
//   lanes 16-31: row = row0+lane-16, K = {k0+8..k0+15, k0+24..k0+31}
__device__ __forceinline__ v16h frag16(const _Float16* __restrict__ base,
                                       int ld, int row0, int k0, int lane) {
  int r  = lane & 15;
  int hi = lane >> 4;
  const _Float16* p = base + (size_t)(row0 + r) * ld + k0 + hi * 8;
  V16U u;
  u.h[0] = *(const v8h*)(p);       // K chunk 0
  u.h[1] = *(const v8h*)(p + 16);  // K chunk 1
  return u.v;
}

__device__ __forceinline__ v8f wmma16(v16h a, v16h b, v8f c) {
  // (neg_a, A, neg_b, B, c_mod, C, reuse_a, reuse_b)
  return __builtin_amdgcn_wmma_f32_16x16x32_f16(false, a, false, b,
                                                (short)0, c, false, false);
}

// ---------------------------------------------------------------------------
// TDM: issue a 1-D tensor_load_to_lds copying `n` f16 elements from `gsrc`
// into LDS byte offset `lds_off`.  D# bit layout per CDNA5 ISA ch.8:
//  group0: [1:0]=count=1, [63:32]=lds_addr, [120:64]=global_addr, [127:126]=2
//  group1: [17:16]=data_size(1=2B), [79:48]=tensor_dim0, [127:112]=tile_dim0,
//          [207:160]=tensor_dim0_stride ; everything else zero (1-D tile)
// This toolchain declares the 6-arg form:
//   (v4u g0, v8i g1, v4i g2, v4i g3, v8i g4, i32 cpol)
// ---------------------------------------------------------------------------
#if __has_builtin(__builtin_amdgcn_tensor_load_to_lds)
#define DTNN_HAS_TDM 1
__device__ __forceinline__ void tdm_load_1d(const _Float16* gsrc,
                                            unsigned lds_off, unsigned n) {
  unsigned long long ga = (unsigned long long)(const void*)gsrc;
  v4u g0;
  g0[0] = 1u;                                   // count=1, user descriptor
  g0[1] = lds_off;                              // lds_addr (bytes)
  g0[2] = (unsigned)(ga & 0xffffffffu);         // global_addr[31:0]
  g0[3] = (unsigned)((ga >> 32) & 0x1ffffffu)   // global_addr[56:32]
        | (2u << 30);                           // type = 2 ("image")
  v8i g1;
  g1[0] = (1 << 16);                            // data_size = 1 -> 2 bytes
  g1[1] = (int)((n & 0xffffu) << 16);           // tensor_dim0[15:0]
  g1[2] = (int)(((n >> 16) & 0xffffu)           // tensor_dim0[31:16]
        | (1u << 16));                          // tensor_dim1 = 1
  g1[3] = (int)((n & 0xffffu) << 16);           // tile_dim0 = n
  g1[4] = 0;                                    // tile_dim1/2 unused
  g1[5] = (int)n;                               // tensor_dim0_stride[31:0]
  g1[6] = 0;
  g1[7] = 0;
  v4i z4 = {0, 0, 0, 0};
  v8i z8 = {0, 0, 0, 0, 0, 0, 0, 0};
  __builtin_amdgcn_tensor_load_to_lds(g0, g1, z4, z4, z8, 0);
}
#else
#define DTNN_HAS_TDM 0
#endif

// ---------------------------------------------------------------------------
// f32 -> f16 conversion (weights + initial activations)
// ---------------------------------------------------------------------------
__global__ void f32_to_f16_kernel(const float* __restrict__ in,
                                  _Float16* __restrict__ out, int n) {
  int i = blockIdx.x * blockDim.x + threadIdx.x;
  if (i < n) out[i] = (_Float16)in[i];
}

// ---------------------------------------------------------------------------
// Streaming reduction: d_sum[b,n,r] = sum_j distance[b,n,j,r]  (268 MB read)
// thread -> (bn, r4) : 16 consecutive lanes cover one row's 64 floats as v4f,
// fully coalesced 128-bit non-temporal loads, f16 output (1 MB).
// Two accumulators break the 128-deep FP add dependency chain.
// ---------------------------------------------------------------------------
__global__ void reduce_dist_kernel(const float* __restrict__ dist,
                                   _Float16* __restrict__ dsum) {
  int idx = blockIdx.x * blockDim.x + threadIdx.x;  // 131072 threads
  int r4  = idx & 15;                               // which v4f in the R=64 row
  int bn  = idx >> 4;                               // 0..8191
  const v4f* p = (const v4f*)(dist + (size_t)bn * 128 * 64) + r4;
  v4f acc0 = {0.f, 0.f, 0.f, 0.f};
  v4f acc1 = {0.f, 0.f, 0.f, 0.f};
#pragma unroll 8
  for (int j = 0; j < 128; j += 2) {
    acc0 += __builtin_nontemporal_load(p + (size_t)j * 16);        // 64-float stride
    acc1 += __builtin_nontemporal_load(p + (size_t)(j + 1) * 16);
  }
  v4f acc = acc0 + acc1;
  _Float16* o = dsum + (size_t)bn * 64 + r4 * 4;
  o[0] = (_Float16)acc[0];
  o[1] = (_Float16)acc[1];
  o[2] = (_Float16)acc[2];
  o[3] = (_Float16)acc[3];
}

// ---------------------------------------------------------------------------
// Fused DTNN layer, all GEMMs transposed (weights = A operand, activations =
// B operand) so every lane's accumulator column is contiguous along the
// feature dim of the row-major buffers -> packed 16B LDS/global stores.
//   cf^T = Wcf @ x^T + bcf          (256x128)(128x16-per-strip)
//   df^T = Wdf @ dsum^T + 128*bdf   (256x64)(64x16)
//   h^T  = Wfc @ p^T                (128x256)(256x16)
//   x'   = h + tanh(h)
// Block = 128 threads = 4 waves; each wave owns one 16-row strip.
// LDS: Wcf(64K)+Wdf(32K)+Wfc(64K) f16 + 4x8K per-wave product = 192 KB.
// Weight staging via the Tensor Data Mover (wave 0 issues 3 D# descriptors,
// DMA engine streams global->LDS, s_wait_tensorcnt + barrier to sync).
// ---------------------------------------------------------------------------
__global__ void __launch_bounds__(128)
dtnn_layer_kernel(const _Float16* __restrict__ xin,    // (8192,128) f16
                  const _Float16* __restrict__ dsum,   // (8192,64)  f16
                  const _Float16* __restrict__ Wcf,    // (256,128)  f16
                  const float*    __restrict__ bcf,    // (256)      f32
                  const _Float16* __restrict__ Wdf,    // (256,64)   f16
                  const float*    __restrict__ bdf,    // (256)      f32
                  const _Float16* __restrict__ Wfc,    // (128,256)  f16
                  _Float16* __restrict__ xout)         // (8192,128) f16
{
  extern __shared__ _Float16 smem[];
  _Float16* sWcf = smem;           // 32768 halves @ LDS byte 0
  _Float16* sWdf = smem + 32768;   // 16384 halves @ byte 65536
  _Float16* sWfc = smem + 49152;   // 32768 halves @ byte 98304
  _Float16* sP   = smem + 81920;   // 4 waves * 16*256 halves (row-major p)

  const int tid = threadIdx.x;

#if DTNN_HAS_TDM
  // Tensor Data Mover staging: one wave issues the three DMA descriptors;
  // dynamic LDS starts at offset 0 (no static __shared__ in this kernel).
  if (tid < 32) {
    tdm_load_1d(Wcf, 0u, 32768u);
    tdm_load_1d(Wdf, 65536u, 16384u);
    tdm_load_1d(Wfc, 98304u, 32768u);
    __builtin_amdgcn_s_wait_tensorcnt(0);
  }
#else
  {
    const v8h* g0 = (const v8h*)Wcf; v8h* s0 = (v8h*)sWcf;
    for (int i = tid; i < 32768 / 8; i += 128) s0[i] = g0[i];
    const v8h* g1 = (const v8h*)Wdf; v8h* s1 = (v8h*)sWdf;
    for (int i = tid; i < 16384 / 8; i += 128) s1[i] = g1[i];
    const v8h* g2 = (const v8h*)Wfc; v8h* s2 = (v8h*)sWfc;
    for (int i = tid; i < 32768 / 8; i += 128) s2[i] = g2[i];
  }
#endif
  __syncthreads();

  const int lane = tid & 31;
  const int wave = tid >> 5;
  const int r    = lane & 15;
  const int hi   = lane >> 4;
  _Float16* sPw  = sP + wave * (16 * 256);

  const int strip = blockIdx.x * 4 + wave;  // 512 strips total
  const int m0    = strip * 16;

  // B-operand fragments (activation strip) are invariant across all H tiles:
  v16h xf[4];
#pragma unroll
  for (int kk = 0; kk < 4; ++kk) xf[kk] = frag16(xin, 128, m0, kk * 32, lane);
  v16h dsf[2];
#pragma unroll
  for (int kk = 0; kk < 2; ++kk) dsf[kk] = frag16(dsum, 64, m0, kk * 32, lane);

  // ---- phase 1: p = (x@Wcf^T + bcf) * (dsum@Wdf^T + 128*bdf), per H-tile ---
  // D tile = [h_local (rows, VGPR idx), m_local (cols, lane)]
  // cf uses two partial accumulators -> zero WMMA->WMMA RAW hazards.
  for (int nt = 0; nt < 16; ++nt) {
    const int n0 = nt * 16;

    // bias along VGPR index = contiguous v8f load
    v8f cacc0 = *(const v8f*)(bcf + n0 + hi * 8);
    v8f cacc1;
#pragma unroll
    for (int i = 0; i < 8; ++i) cacc1[i] = 0.f;
    v8f dacc;
    {
      v8f bd = *(const v8f*)(bdf + n0 + hi * 8);
#pragma unroll
      for (int i = 0; i < 8; ++i) dacc[i] = 128.0f * bd[i];  // N * Wdf_b
    }

    {
      v16h ac = frag16(sWcf, 128, n0, 0, lane);
      cacc0 = wmma16(ac, xf[0], cacc0);
      v16h ad = frag16(sWdf, 64, n0, 0, lane);
      dacc = wmma16(ad, dsf[0], dacc);
      ac = frag16(sWcf, 128, n0, 32, lane);
      cacc1 = wmma16(ac, xf[1], cacc1);
      ad = frag16(sWdf, 64, n0, 32, lane);
      dacc = wmma16(ad, dsf[1], dacc);
      ac = frag16(sWcf, 128, n0, 64, lane);
      cacc0 = wmma16(ac, xf[2], cacc0);
      ac = frag16(sWcf, 128, n0, 96, lane);
      cacc1 = wmma16(ac, xf[3], cacc1);
    }

    // lane holds column m = r; its 8 values are contiguous along H in
    // row-major p -> one packed 16B ds_store
    v8h pp;
#pragma unroll
    for (int i = 0; i < 8; ++i)
      pp[i] = (_Float16)((cacc0[i] + cacc1[i]) * dacc[i]);
    *(v8h*)(sPw + r * 256 + n0 + hi * 8) = pp;
  }

  // ---- phase 2: h^T = Wfc @ p^T ; x' = h + tanh(h) ----
  // K outermost, 8 independent accumulators (one per F tile): no WMMA->WMMA
  // RAW hazard, one shared B fragment per K step.
  v8f hacc[8];
#pragma unroll
  for (int ft = 0; ft < 8; ++ft)
#pragma unroll
    for (int i = 0; i < 8; ++i) hacc[ft][i] = 0.f;

  for (int k = 0; k < 256; k += 32) {
    v16h bp = frag16(sPw, 256, 0, k, lane);
#pragma unroll
    for (int ft = 0; ft < 8; ++ft) {
      v16h a = frag16(sWfc, 256, ft * 16, k, lane);
      hacc[ft] = wmma16(a, bp, hacc[ft]);
    }
  }

  // lane holds column m = r; rows f contiguous -> one packed 16B store/tile
#pragma unroll
  for (int ft = 0; ft < 8; ++ft) {
    v8h pk;
#pragma unroll
    for (int i = 0; i < 8; ++i) {
      float hv = hacc[ft][i];
      pk[i] = (_Float16)(hv + tanhf(hv));
    }
    *(v8h*)(xout + (size_t)(m0 + r) * 128 + ft * 16 + hi * 8) = pk;
  }
}

// ---------------------------------------------------------------------------
// g = sum_n x[b,n,:]; g = g@fc0_w^T + fc0_b; out = g@out_w^T + out_b
// one block per batch element, 256 threads
// ---------------------------------------------------------------------------
__global__ void pool_fc_kernel(const _Float16* __restrict__ x,   // (64,128,128)
                               const float* __restrict__ fc0_w,  // (256,128)
                               const float* __restrict__ fc0_b,  // (256)
                               const float* __restrict__ out_w,  // (1,256)
                               const float* __restrict__ out_b,  // (1)
                               float* __restrict__ out)          // (64)
{
  __shared__ float g[128];
  __shared__ float red[256];
  const int b = blockIdx.x, t = threadIdx.x;

  if (t < 128) {
    float s = 0.f;
    const _Float16* p = x + (size_t)b * 128 * 128 + t;
    for (int n = 0; n < 128; ++n) s += (float)p[n * 128];
    g[t] = s;
  }
  __syncthreads();

  float s = fc0_b[t];
  const float* wrow = fc0_w + t * 128;
  for (int f = 0; f < 128; ++f) s += g[f] * wrow[f];
  red[t] = s * out_w[t];
  __syncthreads();

  for (int k = 128; k > 0; k >>= 1) {
    if (t < k) red[t] += red[t + k];
    __syncthreads();
  }
  if (t == 0) out[b] = red[0] + out_b[0];
}

// ---------------------------------------------------------------------------
extern "C" void kernel_launch(void* const* d_in, const int* in_sizes, int n_in,
                              void* d_out, int out_size, void* d_ws, size_t ws_size,
                              hipStream_t stream) {
  const float* x     = (const float*)d_in[0];
  const float* dist  = (const float*)d_in[1];
  const float* Wcf_w = (const float*)d_in[2];
  const float* Wcf_b = (const float*)d_in[3];
  const float* Wdf_w = (const float*)d_in[4];
  const float* Wdf_b = (const float*)d_in[5];
  const float* Wfc_w = (const float*)d_in[6];
  const float* fc0_w = (const float*)d_in[7];
  const float* fc0_b = (const float*)d_in[8];
  const float* out_w = (const float*)d_in[9];
  const float* out_b = (const float*)d_in[10];

  char* ws = (char*)d_ws;
  _Float16* xA  = (_Float16*)(ws + 0);        // 8192*128 f16 = 2 MB
  _Float16* xB  = (_Float16*)(ws + 2097152);  // 2 MB
  _Float16* dh  = (_Float16*)(ws + 4194304);  // 8192*64 f16 = 1 MB
  _Float16* wcf = (_Float16*)(ws + 5242880);  // 3*256*128 f16
  _Float16* wdf = (_Float16*)(ws + 5439488);  // 3*256*64  f16
  _Float16* wfc = (_Float16*)(ws + 5537792);  // 3*128*256 f16

  // conversions
  f32_to_f16_kernel<<<(1048576 + 255) / 256, 256, 0, stream>>>(x, xA, 1048576);
  f32_to_f16_kernel<<<(98304 + 255) / 256, 256, 0, stream>>>(Wcf_w, wcf, 98304);
  f32_to_f16_kernel<<<(49152 + 255) / 256, 256, 0, stream>>>(Wdf_w, wdf, 49152);
  f32_to_f16_kernel<<<(98304 + 255) / 256, 256, 0, stream>>>(Wfc_w, wfc, 98304);

  // dominant: 268 MB streaming reduction
  reduce_dist_kernel<<<512, 256, 0, stream>>>(dist, dh);

  // 3 fused WMMA layers, ping-pong activations
  _Float16* xin = xA;
  _Float16* xout = xB;
  for (int l = 0; l < 3; ++l) {
    dtnn_layer_kernel<<<128, 128, 196608, stream>>>(
        xin, dh,
        wcf + l * 32768, Wcf_b + l * 256,
        wdf + l * 16384, Wdf_b + l * 256,
        wfc + l * 32768, xout);
    _Float16* t = xin; xin = xout; xout = t;
  }

  pool_fc_kernel<<<64, 256, 0, stream>>>(xin, fc0_w, fc0_b, out_w, out_b,
                                         (float*)d_out);
}
